// MultiHeadAttention_87763361726787
// MI455X (gfx1250) — matrix operand order repaired and checked by reading, not compiled, not run on hardware
//
#include <hip/hip_runtime.h>
#include <hip/hip_bf16.h>

// ---------------------------------------------------------------------------
// MI455X (gfx1250) multi-head-attention (faithful to the reference's
// heads-axis attention + reshape quirk).
//
// Pipeline (all on `stream`):
//   1) fp32 -> bf16 conversion of x, w_qkv, w_out into workspace
//   2) GEMM1: qkv = x * w_qkv^T + b_qkv          (bf16 WMMA, bf16 out)
//   3) attn : per-(n,s) 16x16 heads-attention    (bf16 WMMA + shfl softmax)
//   4) GEMM3: out = attn * w_out^T + b_out       (bf16 WMMA, fp32 out)
//
// GEMM uses LDS double-buffered K-slabs (A 64x32 + B 256x32, 40 KB LDS),
// staged via (best available): TDM tensor_load_to_lds -> async-to-LDS loads
// -> sync global->LDS copies.
//
// Workspace layout (bytes), total ~92.3 MB:
//   [0)          xb     bf16 x       8192*1024*2 = 16,777,216
//   [16777216)   wqkvb  bf16 w_qkv   3072*1024*2 =  6,291,456
//   [23068672)   woutb  bf16 w_out   1024*1024*2 =  2,097,152
//   [25165824)   qkvb   bf16 qkv     8192*3072*2 = 50,331,648
//   [75497472)   attnb  bf16 attnout 8192*1024*2 = 16,777,216
// ---------------------------------------------------------------------------

typedef __attribute__((ext_vector_type(16))) __bf16    v16bf;
typedef __attribute__((ext_vector_type(8)))  float     v8f;
typedef __attribute__((ext_vector_type(4)))  unsigned  v4u;
typedef __attribute__((ext_vector_type(8)))  int       v8i;
typedef __attribute__((ext_vector_type(4)))  int       v4i;

struct __align__(16) U128 { unsigned x, y, z, w; };
union FragU { U128 u[2]; v16bf v; };

// ---- staging-mode selection (all compile-time, guarded) -------------------
#if __has_builtin(__builtin_amdgcn_tensor_load_to_lds) && \
    __has_builtin(__builtin_amdgcn_s_wait_tensorcnt)
#define STAGE_MODE 2   // Tensor Data Mover
#elif __has_builtin(__builtin_amdgcn_global_load_async_to_lds_b128) && \
      __has_builtin(__builtin_amdgcn_s_wait_asynccnt)
#define STAGE_MODE 1   // async global->LDS
#else
#define STAGE_MODE 0   // sync fallback
#endif

__device__ __forceinline__ unsigned short f32_to_bf16_bits(float f) {
  union { float f; unsigned u; } x; x.f = f;
  unsigned r = x.u + 0x7FFFu + ((x.u >> 16) & 1u);
  return (unsigned short)(r >> 16);
}

// A-fragment (16x32 bf16, MxK): lane m = lane&15;
// elems 0..15 = K c0..c0+15, c0 = (lane>>4)*16.
__device__ __forceinline__ v16bf load_frag_a_g(const unsigned short* base,
                                               int ldk, int lane) {
  FragU r;
  const unsigned short* p =
      base + (size_t)(lane & 15) * ldk + ((lane >> 4) * 16);
  r.u[0] = *(const U128*)(p);
  r.u[1] = *(const U128*)(p + 8);
  return r.v;
}

// B-fragment (32x16 bf16, KxN) from B^T rows (K contiguous):
// lane column n = lane&15; K = (lane>>4)*16 + 0..15.
__device__ __forceinline__ v16bf load_frag_b_g(const unsigned short* base,
                                               int ldk, int lane) {
  FragU r;
  const unsigned short* p =
      base + (size_t)(lane & 15) * ldk + ((lane >> 4) * 16);
  r.u[0] = *(const U128*)(p);
  r.u[1] = *(const U128*)(p + 8);
  return r.v;
}

#if STAGE_MODE == 2
// Issue one TDM 2-D tile load: tile_rows x 32 bf16 elements, row stride = K
// elements, dense into LDS at lds_dst (row-major, 32 elems/row).
__device__ __forceinline__ void tdm_load_tile_2d(const unsigned short* gsrc,
                                                 const unsigned short* lds_dst,
                                                 unsigned tile_rows,
                                                 unsigned K) {
  unsigned long long ga = (unsigned long long)gsrc;
  unsigned ldsoff = (unsigned)(unsigned long long)
      (__attribute__((address_space(3))) const unsigned short*)lds_dst;
  v4u g0;
  g0[0] = 1u;                                   // count=1 (valid user D#)
  g0[1] = ldsoff;                               // lds_addr (bytes)
  g0[2] = (unsigned)(ga & 0xFFFFFFFFu);         // global_addr[31:0]
  g0[3] = ((unsigned)(ga >> 32) & 0x01FFFFFFu)  // global_addr[56:32]
          | (2u << 30);                         // type=2 ("image")
  v8i g1;
  g1[0] = (int)0x00010000u;                         // data_size=1 -> 2 bytes
  g1[1] = (int)((K & 0xFFFFu) << 16);               // tensor_dim0[15:0]
  g1[2] = (int)(((K >> 16) & 0xFFFFu)               // tensor_dim0[31:16]
          | ((tile_rows & 0xFFFFu) << 16));         // tensor_dim1[15:0]
  g1[3] = (int)(((tile_rows >> 16) & 0xFFFFu)       // tensor_dim1[31:16]
          | (32u << 16));                           // tile_dim0 = 32
  g1[4] = (int)(tile_rows & 0xFFFFu);               // tile_dim1; tile_dim2=0
  g1[5] = (int)K;                                   // tensor_dim0_stride lo32
  g1[6] = 0;                                        // stride hi16; dim1_stride=0
  g1[7] = 0;
  v4i z4 = {};
#if __has_include(<hip/amd_detail/amd_gfx1250_TDM.h>)
  v8i z8 = {};
  __builtin_amdgcn_tensor_load_to_lds(g0, g1, z4, z4, z8, 0);   // 6-arg form
#else
  __builtin_amdgcn_tensor_load_to_lds(g0, g1, z4, z4, 0);       // 5-arg form
#endif
}
#endif

// Stage one K-slab: A tile 64x32 and B tile 256x32 into LDS.
__device__ __forceinline__ void stage_issue(const unsigned short* Ag,
                                            const unsigned short* Wg, int K,
                                            int k, unsigned short* sAb,
                                            unsigned short* sBb, int tid,
                                            int wave) {
#if STAGE_MODE == 2
  (void)tid;
  if (wave == 0) {
    tdm_load_tile_2d(Ag + k, sAb, 64u, (unsigned)K);
    tdm_load_tile_2d(Wg + k, sBb, 256u, (unsigned)K);
  }
#elif STAGE_MODE == 1
  (void)wave;
  {  // A: thread -> row=tid>>2, 16B chunk=tid&3
    const unsigned short* g = Ag + (size_t)(tid >> 2) * K + k + (tid & 3) * 8;
    unsigned short* l = sAb + (tid >> 2) * 32 + (tid & 3) * 8;
    __builtin_amdgcn_global_load_async_to_lds_b128(
        (__attribute__((address_space(1))) void*)g,
        (__attribute__((address_space(3))) void*)l, 0, 0);
  }
  const unsigned short* g = Wg + (size_t)tid * K + k;  // B: one row per thread
  unsigned short* l = sBb + tid * 32;
#pragma unroll
  for (int c = 0; c < 4; ++c)
    __builtin_amdgcn_global_load_async_to_lds_b128(
        (__attribute__((address_space(1))) void*)(g + c * 8),
        (__attribute__((address_space(3))) void*)(l + c * 8), 0, 0);
#else
  (void)wave;
  {
    const unsigned short* g = Ag + (size_t)(tid >> 2) * K + k + (tid & 3) * 8;
    unsigned short* l = sAb + (tid >> 2) * 32 + (tid & 3) * 8;
    if (k + 32 < K) __builtin_prefetch(g + 32, 0, 3);
    *(U128*)l = *(const U128*)g;
  }
  const unsigned short* g2 = Wg + (size_t)tid * K + k;
  unsigned short* l2 = sBb + tid * 32;
  if (k + 32 < K) __builtin_prefetch(g2 + 32, 0, 3);
#pragma unroll
  for (int c = 0; c < 4; ++c)
    *(U128*)(l2 + c * 8) = *(const U128*)(g2 + c * 8);
#endif
}

__device__ __forceinline__ void stage_wait(int pending, int wave) {
#if STAGE_MODE == 2
  if (wave == 0) {
    if (pending) __builtin_amdgcn_s_wait_tensorcnt(2);
    else         __builtin_amdgcn_s_wait_tensorcnt(0);
  }
#elif STAGE_MODE == 1
  (void)wave;
  if (pending) __builtin_amdgcn_s_wait_asynccnt(5);
  else         __builtin_amdgcn_s_wait_asynccnt(0);
#else
  (void)pending; (void)wave;
#endif
}

// ---------------------------------------------------------------------------
// fp32 -> bf16 converter, 4 elems/thread
// ---------------------------------------------------------------------------
__global__ __launch_bounds__(256) void cvt_f32_bf16_v4(
    const float4* __restrict__ in, uint2* __restrict__ out, int n4) {
  int i = blockIdx.x * 256 + threadIdx.x;
  if (i < n4) {
    float4 f = in[i];
    unsigned lo = (unsigned)f32_to_bf16_bits(f.x) |
                  ((unsigned)f32_to_bf16_bits(f.y) << 16);
    unsigned hi = (unsigned)f32_to_bf16_bits(f.z) |
                  ((unsigned)f32_to_bf16_bits(f.w) << 16);
    out[i] = make_uint2(lo, hi);
  }
}

// ---------------------------------------------------------------------------
// bf16 WMMA GEMM:  C(MxN) = A(MxK) * W(NxK)^T + bias
// Block = 256 threads = 8 waves (2x4); wave tile 32x64; block tile 64x256.
// K-slabs of 32 double-buffered in LDS. M%64==0, N%256==0, K%32==0.
// ---------------------------------------------------------------------------
__global__ __launch_bounds__(256) void gemm_bf16_wmma(
    const unsigned short* __restrict__ A, const unsigned short* __restrict__ W,
    const float* __restrict__ bias, float* __restrict__ outF,
    unsigned short* __restrict__ outB, int M, int N, int K, int writeF) {
  __shared__ __align__(16) unsigned short sA[2][64 * 32];   //  8 KB
  __shared__ __align__(16) unsigned short sB[2][256 * 32];  // 32 KB

  const int tid = threadIdx.x;
  const int lane = tid & 31;
  const int wave = tid >> 5;
  const int wr = wave >> 2, wc = wave & 3;
  const int mBase = blockIdx.y * 64 + wr * 32;
  const int nBase = blockIdx.x * 256 + wc * 64;

  const unsigned short* Ag = A + (size_t)(blockIdx.y * 64) * K;
  const unsigned short* Wg = W + (size_t)(blockIdx.x * 256) * K;

  v8f acc[2][4] = {};

  stage_issue(Ag, Wg, K, 0, sA[0], sB[0], tid, wave);
  const int KT = K >> 5;
  for (int kt = 0; kt < KT; ++kt) {
    const int cur = kt & 1;
    const int pending = (kt + 1 < KT);
    if (pending)
      stage_issue(Ag, Wg, K, (kt + 1) << 5, sA[cur ^ 1], sB[cur ^ 1], tid, wave);
    stage_wait(pending, wave);
    __syncthreads();  // slab `cur` visible to all waves

    const unsigned short* aT = sA[cur] + (wr * 32) * 32;
    const unsigned short* bT = sB[cur] + (wc * 64) * 32;
    v16bf a0 = load_frag_a_g(aT, 32, lane);
    v16bf a1 = load_frag_a_g(aT + 16 * 32, 32, lane);
    v16bf b0 = load_frag_b_g(bT + 0 * 512, 32, lane);
    v16bf b1 = load_frag_b_g(bT + 1 * 512, 32, lane);
    v16bf b2 = load_frag_b_g(bT + 2 * 512, 32, lane);
    v16bf b3 = load_frag_b_g(bT + 3 * 512, 32, lane);
    acc[0][0] = __builtin_amdgcn_wmma_f32_16x16x32_bf16(false, a0, false, b0,
                (short)0, acc[0][0], false, false);
    acc[0][1] = __builtin_amdgcn_wmma_f32_16x16x32_bf16(false, a0, false, b1,
                (short)0, acc[0][1], false, false);
    acc[0][2] = __builtin_amdgcn_wmma_f32_16x16x32_bf16(false, a0, false, b2,
                (short)0, acc[0][2], false, false);
    acc[0][3] = __builtin_amdgcn_wmma_f32_16x16x32_bf16(false, a0, false, b3,
                (short)0, acc[0][3], false, false);
    acc[1][0] = __builtin_amdgcn_wmma_f32_16x16x32_bf16(false, a1, false, b0,
                (short)0, acc[1][0], false, false);
    acc[1][1] = __builtin_amdgcn_wmma_f32_16x16x32_bf16(false, a1, false, b1,
                (short)0, acc[1][1], false, false);
    acc[1][2] = __builtin_amdgcn_wmma_f32_16x16x32_bf16(false, a1, false, b2,
                (short)0, acc[1][2], false, false);
    acc[1][3] = __builtin_amdgcn_wmma_f32_16x16x32_bf16(false, a1, false, b3,
                (short)0, acc[1][3], false, false);
    __syncthreads();  // all reads of slab `cur` done before it is re-filled
  }

  // C layout: VGPR r -> M = r + (lane>>4)*8, N = lane&15.
  const int lh = lane >> 4, ln = lane & 15;
#pragma unroll
  for (int j = 0; j < 4; j++) {
    const int n = nBase + j * 16 + ln;
    const float bj = bias[n];
#pragma unroll
    for (int i = 0; i < 2; i++) {
#pragma unroll
      for (int r = 0; r < 8; r++) {
        const int m = mBase + i * 16 + lh * 8 + r;
        const float v = acc[i][j][r] + bj;
        if (writeF)
          outF[(size_t)m * N + n] = v;
        else
          outB[(size_t)m * N + n] = f32_to_bf16_bits(v);
      }
    }
  }
}

// ---------------------------------------------------------------------------
// Heads-axis attention, one wave per (batch, position).
// qkv row layout: [Q(16h x 64d) | K(16h x 64d) | V(16h x 64d)] = 3072 bf16.
// energy(16x16) = Q * K^T (2 WMMA); softmax over k-head axis (shfl_xor);
// out(16x64) = P * V (4 WMMA, Kdim 16 zero-padded to 32; P via LDS, V^T LDS).
// Stored in the reference's reshape layout: attnb[n, q*128+s/16, (s%16)*64+d]
// ---------------------------------------------------------------------------
__global__ __launch_bounds__(256) void attn_heads_wmma(
    const unsigned short* __restrict__ qkv, unsigned short* __restrict__ attnb) {
  __shared__ __align__(16) unsigned short vt[8][64 * 16];  // V^T per wave
  __shared__ __align__(16) unsigned short pp[8][16 * 32];  // P (padded) per wave

  const int lane = threadIdx.x & 31;
  const int wave = threadIdx.x >> 5;
  const int lh = lane >> 4, ln = lane & 15;
  const int idx = blockIdx.x * 8 + wave;  // n*2048 + s
  const int nb = idx >> 11;
  const int s = idx & 2047;
  const unsigned short* base = qkv + (size_t)idx * 3072;

  unsigned short* vts = vt[wave];
  unsigned short* pps = pp[wave];

  for (int i = lane; i < 256; i += 32) {  // zero P pad cols 16..31
    int m = i >> 4, c = i & 15;
    pps[m * 32 + 16 + c] = 0;
  }
  for (int i = lane; i < 1024; i += 32) {  // stage V^T (64x16)
    int h = i >> 6, d = i & 63;
    vts[d * 16 + h] = base[2048 + h * 64 + d];
  }

  v8f e = {};
#pragma unroll
  for (int k = 0; k < 64; k += 32) {
    v16bf a = load_frag_a_g(base + k, 64, lane);         // Q rows = heads
    v16bf b = load_frag_b_g(base + 1024 + k, 64, lane);  // K rows = columns
    e = __builtin_amdgcn_wmma_f32_16x16x32_bf16(false, a, false, b, (short)0,
                                                e, false, false);
  }

#pragma unroll
  for (int r = 0; r < 8; r++) {  // softmax over k-head axis (16 lanes)
    float t = e[r] * 0.125f;     // 1/sqrt(64)
    float mx = t;
    mx = fmaxf(mx, __shfl_xor(mx, 1));
    mx = fmaxf(mx, __shfl_xor(mx, 2));
    mx = fmaxf(mx, __shfl_xor(mx, 4));
    mx = fmaxf(mx, __shfl_xor(mx, 8));
    float ex = __expf(t - mx);
    float sm = ex;
    sm += __shfl_xor(sm, 1);
    sm += __shfl_xor(sm, 2);
    sm += __shfl_xor(sm, 4);
    sm += __shfl_xor(sm, 8);
    e[r] = ex / sm;
  }

#pragma unroll
  for (int r = 0; r < 8; r++)  // P (C-layout) -> LDS row-major 16x32
    pps[(lh * 8 + r) * 32 + ln] = f32_to_bf16_bits(e[r]);

  __syncthreads();

  FragU pa;  // A-fragment of P (upper K half zero)
  {
    const unsigned short* p = pps + (size_t)ln * 32 + lh * 16;
    pa.u[0] = *(const U128*)(p);
    pa.u[1] = *(const U128*)(p + 8);
  }

#pragma unroll
  for (int t = 0; t < 4; t++) {  // out = P * V, 4 N-tiles over d
    FragU vb;
    if (lh == 0) {
      const unsigned short* p = vts + (size_t)(t * 16 + ln) * 16;
      vb.u[0] = *(const U128*)(p);
      vb.u[1] = *(const U128*)(p + 8);
    } else {
      vb.u[0] = U128{0u, 0u, 0u, 0u};
      vb.u[1] = U128{0u, 0u, 0u, 0u};
    }
    v8f o = {};
    o = __builtin_amdgcn_wmma_f32_16x16x32_bf16(false, pa.v, false, vb.v,
                                                (short)0, o, false, false);
#pragma unroll
    for (int r = 0; r < 8; r++) {
      const int q = lh * 8 + r;
      const int d = t * 16 + ln;
      const size_t off =
          ((size_t)nb * 2048 + (size_t)q * 128 + (s >> 4)) * 1024 +
          (size_t)(s & 15) * 64 + d;
      attnb[off] = f32_to_bf16_bits(o[r]);
    }
  }
}

// ---------------------------------------------------------------------------
extern "C" void kernel_launch(void* const* d_in, const int* in_sizes, int n_in,
                              void* d_out, int out_size, void* d_ws,
                              size_t ws_size, hipStream_t stream) {
  (void)in_sizes; (void)n_in; (void)out_size; (void)ws_size;
  const float* x     = (const float*)d_in[0];  // (4,2048,1024)
  const float* w_qkv = (const float*)d_in[1];  // (3072,1024)
  const float* b_qkv = (const float*)d_in[2];  // (3072,)
  const float* w_out = (const float*)d_in[3];  // (1024,1024)
  const float* b_out = (const float*)d_in[4];  // (1024,)
  float* out = (float*)d_out;                  // (4,2048,1024)

  char* ws = (char*)d_ws;
  unsigned short* xb    = (unsigned short*)(ws);
  unsigned short* wqkvb = (unsigned short*)(ws + 16777216);
  unsigned short* woutb = (unsigned short*)(ws + 23068672);
  unsigned short* qkvb  = (unsigned short*)(ws + 25165824);
  unsigned short* attnb = (unsigned short*)(ws + 75497472);

  cvt_f32_bf16_v4<<<8192, 256, 0, stream>>>((const float4*)x, (uint2*)xb,
                                            2097152);
  cvt_f32_bf16_v4<<<3072, 256, 0, stream>>>((const float4*)w_qkv,
                                            (uint2*)wqkvb, 786432);
  cvt_f32_bf16_v4<<<1024, 256, 0, stream>>>((const float4*)w_out,
                                            (uint2*)woutb, 262144);

  gemm_bf16_wmma<<<dim3(12, 128), 256, 0, stream>>>(
      xb, wqkvb, b_qkv, nullptr, qkvb, 8192, 3072, 1024, 0);

  attn_heads_wmma<<<1024, 256, 0, stream>>>(qkvb, attnb);

  gemm_bf16_wmma<<<dim3(4, 128), 256, 0, stream>>>(
      attnb, woutb, b_out, out, nullptr, 8192, 1024, 1024, 1);
}
